// ContrastiveLoss_11072425689236
// MI455X (gfx1250) — matrix operand order repaired and checked by reading, not compiled, not run on hardware
//
#include <hip/hip_runtime.h>
#include <hip/hip_bf16.h>
#include <math.h>
#include <stdint.h>

typedef __attribute__((ext_vector_type(16))) __bf16 v16bf;
typedef __attribute__((ext_vector_type(8)))  __bf16 v8bf;
typedef __attribute__((ext_vector_type(4)))  __bf16 v4bf;
typedef __attribute__((ext_vector_type(8)))  float  v8f;

#define N_    16384
#define D_    512
#define BM    128
#define BN    64
#define PITCH 520   // 512 + 8 bf16 pad: 1040B row stride -> rows land 4 banks apart
#define A_BYTES   ((size_t)BM * PITCH * 2)          // 133120
#define B_BYTES   ((size_t)BN * PITCH * 2)          // 66560
#define SMEM_BYTES (A_BYTES + 2 * B_BYTES)          // 266240 <= 320K

// ---- CDNA5 async global->LDS copy (ASYNCcnt-tracked DMA, no staging VGPRs) ----
__device__ __forceinline__ unsigned lds_addr_of(const void* p) {
  return (unsigned)(uintptr_t)(__attribute__((address_space(3))) const char*)p;
}
__device__ __forceinline__ void async_b128(unsigned lds_off, const void* gptr) {
  asm volatile("global_load_async_to_lds_b128 %0, %1, off"
               :: "v"(lds_off), "v"((unsigned long long)(uintptr_t)gptr)
               : "memory");
}
__device__ __forceinline__ void wait_async0() {
  asm volatile("s_wait_asynccnt 0" ::: "memory");
}

// ---------------------------------------------------------------------------
// 1) fp32 -> bf16 conversion; logit_scale folded into image features.
// ---------------------------------------------------------------------------
__global__ void clip_convert_kernel(const float* __restrict__ img,
                                    const float* __restrict__ txt,
                                    const float* __restrict__ scale_p,
                                    __bf16* __restrict__ imgS,
                                    __bf16* __restrict__ txtb) {
  const float sc = scale_p[0];
  size_t i = (size_t)blockIdx.x * blockDim.x + threadIdx.x;   // one float4 chunk
  float4 vi = ((const float4*)img)[i];
  float4 vt = ((const float4*)txt)[i];
  v4bf ri = { (__bf16)(vi.x * sc), (__bf16)(vi.y * sc),
              (__bf16)(vi.z * sc), (__bf16)(vi.w * sc) };
  v4bf rt = { (__bf16)vt.x, (__bf16)vt.y, (__bf16)vt.z, (__bf16)vt.w };
  ((v4bf*)imgS)[i] = ri;
  ((v4bf*)txtb)[i] = rt;
}

// ---------------------------------------------------------------------------
// 2) diag[n] = dot(imgS[n,:], txt[n,:])  (one wave per row)
// ---------------------------------------------------------------------------
__global__ void clip_diag_kernel(const __bf16* __restrict__ a,
                                 const __bf16* __restrict__ b,
                                 float* __restrict__ diag) {
  const int lane = threadIdx.x & 31;
  const int row  = blockIdx.x * (blockDim.x >> 5) + (threadIdx.x >> 5);
  const __bf16* ar = a + (size_t)row * D_ + lane * 16;
  const __bf16* br = b + (size_t)row * D_ + lane * 16;
  float s = 0.f;
#pragma unroll
  for (int c = 0; c < 2; ++c) {
    v8bf av = *(const v8bf*)(ar + c * 8);
    v8bf bv = *(const v8bf*)(br + c * 8);
#pragma unroll
    for (int e = 0; e < 8; ++e) s += (float)av[e] * (float)bv[e];
  }
#pragma unroll
  for (int d = 1; d < 32; d <<= 1) s += __shfl_xor(s, d, 32);
  if (lane == 0) diag[row] = s;
}

// ---------------------------------------------------------------------------
// 3) Fused GEMM + online logsumexp over rows of  A @ B^T  (A,B: [N,D] bf16).
//    blockIdx.y==0: A=imgS,B=txt  -> lse_i ; blockIdx.y==1: swapped -> lse_t.
//    A block pinned in LDS; B tiles double-buffered via async-to-LDS DMA.
// ---------------------------------------------------------------------------
__global__ __launch_bounds__(256)
void clip_lse_kernel(const __bf16* __restrict__ imgS,
                     const __bf16* __restrict__ txtb,
                     float* __restrict__ lse_i,
                     float* __restrict__ lse_t) {
  extern __shared__ char smem[];
  __bf16* Al  = (__bf16*)smem;                       // BM x PITCH
  __bf16* Bb0 = (__bf16*)(smem + A_BYTES);           // BN x PITCH (ping)
  __bf16* Bb1 = (__bf16*)(smem + A_BYTES + B_BYTES); // BN x PITCH (pong)

  const bool tpass = (blockIdx.y != 0);
  const __bf16* A = tpass ? txtb : imgS;
  const __bf16* B = tpass ? imgS : txtb;
  float* lse = tpass ? lse_t : lse_i;

  const int tid   = threadIdx.x;
  const int lane  = tid & 31;
  const int wid   = tid >> 5;          // 8 waves, each owns 16 rows
  const int hs    = (lane < 16) ? 0 : 1;
  const int rbase = blockIdx.x * BM;

  // ---- issue async DMA: A block (32 chunks/thread) ----
  {
    const unsigned abase = lds_addr_of(Al);
#pragma unroll 4
    for (int it = 0; it < (BM * D_ / 8) / 256; ++it) {
      int cidx = tid + it * 256;
      int r  = cidx >> 6;                // D/8 = 64 chunks per row
      int c8 = (cidx & 63) * 8;
      async_b128(abase + (unsigned)(r * PITCH + c8) * 2,
                 A + (size_t)(rbase + r) * D_ + c8);
    }
  }
  // ---- issue async DMA: B tile 0 into ping buffer (16 chunks/thread) ----
  {
    const unsigned bbase = lds_addr_of(Bb0);
#pragma unroll 4
    for (int it = 0; it < (BN * D_ / 8) / 256; ++it) {
      int cidx = tid + it * 256;
      int r  = cidx >> 6;
      int c8 = (cidx & 63) * 8;
      async_b128(bbase + (unsigned)(r * PITCH + c8) * 2,
                 B + (size_t)r * D_ + c8);
    }
  }

  float Mi[8], Si[8];
#pragma unroll
  for (int i = 0; i < 8; ++i) { Mi[i] = -1e30f; Si[i] = 0.f; }

  // A fragment base: lane = M (mod 16); K chunks {k..k+7, k+16..k+23} (lanes<16)
  //                                  or {k+8..k+15, k+24..k+31}       (lanes>=16)
  const __bf16* arow = Al + (wid * 16 + (lane & 15)) * PITCH;

  for (int j = 0; j < N_ / BN; ++j) {
    const __bf16* Bl = (j & 1) ? Bb1 : Bb0;
    wait_async0();                      // this wave's tile-j DMA done
    __syncthreads();                    // all waves' DMA done; prev compute done

    if (j + 1 < N_ / BN) {              // prefetch tile j+1 into the other buffer
      const unsigned bbase = lds_addr_of((j & 1) ? Bb0 : Bb1);
      const __bf16* Bsrc = B + (size_t)(j + 1) * BN * D_;
#pragma unroll 4
      for (int it = 0; it < (BN * D_ / 8) / 256; ++it) {
        int cidx = tid + it * 256;
        int r  = cidx >> 6;
        int c8 = (cidx & 63) * 8;
        async_b128(bbase + (unsigned)(r * PITCH + c8) * 2,
                   Bsrc + (size_t)r * D_ + c8);
      }
    }

    v8f c[4];
#pragma unroll
    for (int n = 0; n < 4; ++n) c[n] = (v8f){0.f,0.f,0.f,0.f,0.f,0.f,0.f,0.f};

#pragma unroll 4
    for (int k = 0; k < D_; k += 32) {
      const int ka = k + hs * 8;
      v8bf a0 = *(const v8bf*)(arow + ka);
      v8bf a1 = *(const v8bf*)(arow + ka + 16);
      v16bf af = __builtin_shufflevector(a0, a1,
                   0,1,2,3,4,5,6,7,8,9,10,11,12,13,14,15);
      const int kb = k + hs * 8;        // B: lane = N; same K chunks as A
      v16bf bq[4];
#pragma unroll
      for (int n = 0; n < 4; ++n) {     // load ALL fragments first -> partial waits
        const __bf16* brow = Bl + (n * 16 + (lane & 15)) * PITCH;
        v8bf b0 = *(const v8bf*)(brow + kb);
        v8bf b1 = *(const v8bf*)(brow + kb + 16);
        bq[n] = __builtin_shufflevector(b0, b1,
                  0,1,2,3,4,5,6,7,8,9,10,11,12,13,14,15);
      }
#pragma unroll
      for (int n = 0; n < 4; ++n)
        c[n] = __builtin_amdgcn_wmma_f32_16x16x32_bf16(
                 false, af, false, bq[n], (short)0, c[n], false, false);
    }

    // ---- per-lane online logsumexp update (no cross-lane traffic here) ----
#pragma unroll
    for (int i = 0; i < 8; ++i) {
      float tmax = c[0][i];
#pragma unroll
      for (int n = 1; n < 4; ++n) tmax = fmaxf(tmax, c[n][i]);
      float nm = fmaxf(Mi[i], tmax);
      float sum = 0.f;
#pragma unroll
      for (int n = 0; n < 4; ++n) sum += __expf(c[n][i] - nm);
      Si[i] = Si[i] * __expf(Mi[i] - nm) + sum;
      Mi[i] = nm;
    }
  }

  // ---- merge the 16 column-lanes of each row, then write lse ----
  float outv[8];
#pragma unroll
  for (int i = 0; i < 8; ++i) {
    float m = Mi[i], s = Si[i];
#pragma unroll
    for (int d = 1; d < 16; d <<= 1) {
      float mo = __shfl_xor(m, d, 32);
      float so = __shfl_xor(s, d, 32);
      float nm = fmaxf(m, mo);
      s = s * __expf(m - nm) + so * __expf(mo - nm);
      m = nm;
    }
    outv[i] = m + __logf(s);
  }
  if ((lane & 15) == 0) {
    const int r0 = rbase + wid * 16 + hs * 8;   // VGPR i -> row i (lo half) / i+8
#pragma unroll
    for (int i = 0; i < 8; ++i) lse[r0 + i] = outv[i];
  }
}

// ---------------------------------------------------------------------------
// 4) loss = 0.5 * mean(lse_i + lse_t - 2*diag)
// ---------------------------------------------------------------------------
__global__ void clip_loss_kernel(const float* __restrict__ lse_i,
                                 const float* __restrict__ lse_t,
                                 const float* __restrict__ diag,
                                 float* __restrict__ out) {
  __shared__ float red[8];
  float s = 0.f;
  for (int n = threadIdx.x; n < N_; n += 256)
    s += lse_i[n] + lse_t[n] - 2.f * diag[n];
#pragma unroll
  for (int d = 1; d < 32; d <<= 1) s += __shfl_xor(s, d, 32);
  if ((threadIdx.x & 31) == 0) red[threadIdx.x >> 5] = s;
  __syncthreads();
  if (threadIdx.x == 0) {
    float t = 0.f;
#pragma unroll
    for (int w = 0; w < 8; ++w) t += red[w];
    out[0] = t * (0.5f / (float)N_);
  }
}

// ---------------------------------------------------------------------------
extern "C" void kernel_launch(void* const* d_in, const int* in_sizes, int n_in,
                              void* d_out, int out_size, void* d_ws, size_t ws_size,
                              hipStream_t stream) {
  const float* img     = (const float*)d_in[0];
  const float* txt     = (const float*)d_in[1];
  const float* scale_p = (const float*)d_in[2];
  float* out = (float*)d_out;

  char* ws = (char*)d_ws;
  __bf16* imgS = (__bf16*)ws;                                 // 16 MB
  __bf16* txtb = (__bf16*)(ws + (size_t)N_ * D_ * 2);         // 16 MB
  float*  lse_i = (float*)(ws + (size_t)N_ * D_ * 4);         // 64 KB
  float*  lse_t = lse_i + N_;                                 // 64 KB
  float*  diag  = lse_t + N_;                                 // 64 KB

  clip_convert_kernel<<<(N_ * D_ / 4) / 256, 256, 0, stream>>>(
      img, txt, scale_p, imgS, txtb);

  clip_diag_kernel<<<N_ / 8, 256, 0, stream>>>(imgS, txtb, diag);

  hipFuncSetAttribute((const void*)clip_lse_kernel,
                      hipFuncAttributeMaxDynamicSharedMemorySize,
                      (int)SMEM_BYTES);
  clip_lse_kernel<<<dim3(N_ / BM, 2), 256, SMEM_BYTES, stream>>>(
      imgS, txtb, lse_i, lse_t);

  clip_loss_kernel<<<1, 256, 0, stream>>>(lse_i, lse_t, diag, out);
}
